// AllegroLayer_54674933678511
// MI455X (gfx1250) — matrix operand order repaired and checked by reading, compiled, not run on hardware
//
#include <hip/hip_runtime.h>
#include <hip/hip_fp16.h>
#include <math.h>
#include <stdint.h>

typedef __attribute__((ext_vector_type(16))) _Float16 v16h;
typedef __attribute__((ext_vector_type(8)))  _Float16 v8h;
typedef __attribute__((ext_vector_type(8)))  float    v8f;
typedef __attribute__((ext_vector_type(4)))  unsigned v4u;
typedef __attribute__((ext_vector_type(4)))  int      v4i;
typedef __attribute__((ext_vector_type(8)))  int      v8i;

#define N_NODES 4096
#define EPS_W   0.25f

// workspace layout (bytes)
#define WS_NODE_BYTES  16777216u                 // nodeAcc: 4096*1024 f32
#define WS_WP_OFF      16777216u                 // packed f16 weights: 69632 halves
#define WS_XH_OFF      (16777216u + 139264u)     // packed f16 x: E*64 halves
// packed-weight sub-offsets (halves)
#define WP_W1   0
#define WP_W2   16384
#define WP_W3   20480
#define WP_WL1  24576
#define WP_WL2  45056
#define WP_WW   65536
#define WP_TOTAL 69632

// ---- exact real-w3j (1,1,2) tensor, unit Frobenius norm (a = 1/sqrt(10)) ----
__constant__ float C112c[3][3][5] = {
  { {0.f,0.f,-0.18257419f,0.f,-0.31622777f},
    {0.f,0.31622777f,0.f,0.f,0.f},
    {0.31622777f,0.f,0.f,0.f,0.f} },
  { {0.f,0.31622777f,0.f,0.f,0.f},
    {0.f,0.f,0.36514837f,0.f,0.f},
    {0.f,0.f,0.f,0.31622777f,0.f} },
  { {0.31622777f,0.f,0.f,0.f,0.f},
    {0.f,0.f,0.f,0.31622777f,0.f},
    {0.f,0.f,-0.18257419f,0.f,0.31622777f} }
};

__device__ __forceinline__ float surro(int l1, int l2, int l3, int i, int j, int k) {
  int m1 = i - l1, m2 = j - l2, m3 = k - l3;
  int d = m1 + m2 - m3;
  float v = 0.9f / (1.0f + (float)(d * d));
  if (((unsigned)(m1 + m2 + m3 + 16)) & 1u) v = -v;
  return v * rsqrtf((float)((2 * l1 + 1) * (2 * l2 + 1) * (2 * l3 + 1)));
}

__device__ __forceinline__ v8f zero8() {
  v8f z;
#pragma unroll
  for (int i = 0; i < 8; ++i) z[i] = 0.0f;
  return z;
}

__device__ __forceinline__ v8f wmma16(v16h a, v16h b, v8f c) {
  return __builtin_amdgcn_wmma_f32_16x16x32_f16(false, a, false, b, (short)0, c, false, false);
}

// A-fragment (16x32 f16), table-order k layout (k = 8h+s, 16+8h+s-8): two 16B loads per lane
__device__ __forceinline__ v16h load_a_c(const _Float16* base, int rowStride, int kt, int lane) {
  int m  = lane & 15;
  int kh = (lane >> 4) * 8;
  const _Float16* p = base + m * rowStride + kt * 32 + kh;
  v8h lo = *(const v8h*)p;
  v8h hi = *(const v8h*)(p + 16);
  v16h a;
#pragma unroll
  for (int i = 0; i < 8; ++i) { a[i] = lo[i]; a[8 + i] = hi[i]; }
  return a;
}

// B-fragment (32x16 f16) from pre-packed fragment-order weights: 32 contiguous bytes/lane
__device__ __forceinline__ v16h load_b_pk(const _Float16* Wp, int nt, int kt, int lane) {
  int n = nt * 16 + (lane & 15), half = lane >> 4;
  const _Float16* p = Wp + (((kt * 64 + n) * 2 + half) << 4);
  v8h lo = *(const v8h*)p;
  v8h hi = *(const v8h*)(p + 8);
  v16h b;
#pragma unroll
  for (int i = 0; i < 8; ++i) { b[i] = lo[i]; b[8 + i] = hi[i]; }
  return b;
}

__device__ __forceinline__ void wave_sync_lds() {
  asm volatile("s_wait_dscnt 0" ::: "memory");
  __builtin_amdgcn_wave_barrier();
}

__device__ __forceinline__ float silu(float v) { return v / (1.0f + __expf(-v)); }

// spherical harmonics up to l=3 for one edge vector
__device__ __forceinline__ void sh16(float vx, float vy, float vz, float* Y) {
  float inv = 1.0f / (sqrtf(vx * vx + vy * vy + vz * vz) + 1e-12f);
  float y1[3] = {1.7320508f * vx * inv, 1.7320508f * vy * inv, 1.7320508f * vz * inv};
  float y2[5];
#pragma unroll
  for (int k = 0; k < 5; ++k) {
    float acc = 0.0f;
#pragma unroll
    for (int i = 0; i < 3; ++i)
#pragma unroll
      for (int j = 0; j < 3; ++j) acc += y1[i] * y1[j] * C112c[i][j][k];
    y2[k] = acc * 2.0412415f;
  }
  Y[0] = 1.0f;
#pragma unroll
  for (int k = 0; k < 3; ++k) Y[1 + k] = y1[k];
#pragma unroll
  for (int k = 0; k < 5; ++k) Y[4 + k] = y2[k];
#pragma unroll
  for (int k = 0; k < 7; ++k) {
    float acc = 0.0f;
#pragma unroll
    for (int i = 0; i < 5; ++i)
#pragma unroll
      for (int j = 0; j < 3; ++j) acc += y2[i] * y1[j] * surro(2, 1, 3, i, j, k);
    Y[9 + k] = acc * 0.8f;
  }
}

// ------------- kernel A: pack all weights to f16 in B-fragment order -------------
__global__ void __launch_bounds__(256)
pack_weights_kernel(const float* __restrict__ W1, const float* __restrict__ W2,
                    const float* __restrict__ W3, const float* __restrict__ Wl1,
                    const float* __restrict__ Wl2, const float* __restrict__ Ww,
                    _Float16* __restrict__ Wp) {
  int gid = blockIdx.x * 256 + threadIdx.x;
  if (gid >= WP_TOTAL) return;
  const float* src;
  int base, rel;
  if      (gid < WP_W2)  { src = W1;  base = WP_W1;  rel = gid - WP_W1;  }
  else if (gid < WP_W3)  { src = W2;  base = WP_W2;  rel = gid - WP_W2;  }
  else if (gid < WP_WL1) { src = W3;  base = WP_W3;  rel = gid - WP_W3;  }
  else if (gid < WP_WL2) { src = Wl1; base = WP_WL1; rel = gid - WP_WL1; }
  else if (gid < WP_WW)  { src = Wl2; base = WP_WL2; rel = gid - WP_WL2; }
  else                   { src = Ww;  base = WP_WW;  rel = gid - WP_WW;  }
  int e = rel & 15, half = (rel >> 4) & 1, n = (rel >> 5) & 63, kt = rel >> 11;
  int k = kt * 32 + half * 8 + e + ((e >> 3) << 3);
  Wp[base + rel] = (_Float16)src[k * 64 + n];
}

// ------------- kernel B: pack x to f16 (A-fragment friendly, row-major) -------------
__global__ void __launch_bounds__(256)
pack_x_kernel(const float* __restrict__ x, _Float16* __restrict__ xh, int n) {
  int gid = blockIdx.x * 256 + threadIdx.x;
  if (gid < n) xh[gid] = (_Float16)x[gid];
}

// ------------- kernel C: w = x@Ww via WMMA, SH, scatter w*Y into nodeAcc -------------
__global__ void __launch_bounds__(256)
sh_scatter_kernel(const float* __restrict__ vectors, const _Float16* __restrict__ xh,
                  const int* __restrict__ senders, const _Float16* __restrict__ Wp,
                  float* __restrict__ nodeAcc, int E) {
  __shared__ float wsh[16][64];
  __shared__ float Ysh[16][16];
  int tid = threadIdx.x, lane = tid & 31;
  int e0 = blockIdx.x * 16;

  if (tid < 32) {  // wave 0: 16x64 = (16x64 xh-tile) @ Ww via WMMA
    v8f acc[4] = {zero8(), zero8(), zero8(), zero8()};
    for (int kt = 0; kt < 2; ++kt) {
      v16h a = load_a_c(xh + (size_t)e0 * 64, 64, kt, lane);
#pragma unroll
      for (int t = 0; t < 4; ++t)
        acc[t] = wmma16(a, load_b_pk(Wp + WP_WW, t, kt, lane), acc[t]);
    }
#pragma unroll
    for (int t = 0; t < 4; ++t)
#pragma unroll
      for (int r = 0; r < 8; ++r) {
        int m = r + (lane >> 4) * 8, n = t * 16 + (lane & 15);
        wsh[m][n] = acc[t][r] * 0.125f;  // 1/sqrt(64)
      }
  } else if (tid >= 32 && tid < 48) {  // 16 threads: SH per edge
    int le = tid - 32;
    int e = e0 + le;
    float Y[16];
    sh16(vectors[e * 3], vectors[e * 3 + 1], vectors[e * 3 + 2], Y);
#pragma unroll
    for (int k = 0; k < 16; ++k) Ysh[le][k] = Y[k];
  }
  __syncthreads();

  // scatter: 16 edges x 64 ch x 16 sh = 16384 atomics, 64 per thread
  for (int t = 0; t < 64; ++t) {
    int flat = tid * 64 + t;
    int le = flat >> 10, rest = flat & 1023, u = rest >> 4, k = rest & 15;
    int s = senders[e0 + le];
    atomicAdd(nodeAcc + (size_t)s * 1024 + u * 16 + k, wsh[le][u] * Ysh[le][k]);
  }
}

// ------------- kernel D: TDM gather + tensor products + WMMA GEMMs -------------
__global__ void __launch_bounds__(256)
allegro_main_kernel(const float* __restrict__ vectors, const _Float16* __restrict__ xh,
                    const float* __restrict__ V, const int* __restrict__ senders,
                    const _Float16* __restrict__ Wp, const float* __restrict__ nodeAcc,
                    float* __restrict__ out, int E) {
  // manual LDS pool: wY region (64KB) is dead after phase 1 and reused for h1/h2
  __shared__ alignas(16) unsigned char smem[157056];
  float*    wYsh  = (float*)smem;                   // [16][1024] f32 (phase 1)
  _Float16* h1    = (_Float16*)smem;                // [16][64]   (phase 2)
  _Float16* h2    = (_Float16*)(smem + 2048);       // [16][64]
  _Float16* Psh   = (_Float16*)(smem + 65536);      // [3][16][320] comp-major
  _Float16* Dsh   = (_Float16*)(smem + 96256);      // [5][16][320]
  _Float16* hin   = (_Float16*)(smem + 147456);     // [16][256]
  float*    envsh = (float*)(smem + 155648);        // [16]
  float*    C321s = (float*)(smem + 155712);        // 105
  float*    C222s = C321s + 105;                    // 125
  float*    C312s = C222s + 125;                    // 105

  int tid = threadIdx.x;
  int e0 = blockIdx.x * 16;

  // ---- TDM: gather 16 rows of nodeAcc (1024 f32 each) selected by senders -> LDS ----
  if (tid < 32) {
    unsigned lds_addr = (unsigned)(uintptr_t)wYsh;  // LDS aperture: low bits = LDS offset
    unsigned long long ga = (unsigned long long)(uintptr_t)nodeAcc;
    v4u g0;
    g0[0] = 1u | (1u << 31);                         // count=1, gather_mode=1, 16-bit idx
    g0[1] = lds_addr;
    g0[2] = (unsigned)(ga & 0xFFFFFFFFu);
    g0[3] = (unsigned)((ga >> 32) & 0x01FFFFFFu) | 0x80000000u;  // addr[56:32] | type=2
    v8i g1;
    g1[0] = 0x00020000;                              // data_size = 4B
    g1[1] = (int)(1024u << 16);                      // tensor_dim0 = 1024
    g1[2] = (int)(4096u << 16);                      // tensor_dim1 = 4096
    g1[3] = (int)(1024u << 16);                      // tile_dim0 = 1024
    g1[4] = 16;                                      // tile_dim1 = #indices
    g1[5] = 1024;                                    // tensor_dim0_stride
    g1[6] = 0; g1[7] = 0;
    v4i g2, g3;
#pragma unroll
    for (int i = 0; i < 4; ++i) {
      g2[i] = (senders[e0 + 2 * i] & 0xFFFF)     | (senders[e0 + 2 * i + 1] << 16);
      g3[i] = (senders[e0 + 8 + 2 * i] & 0xFFFF) | (senders[e0 + 9 + 2 * i] << 16);
    }
#if defined(__clang_major__) && __clang_major__ >= 23
    v8i g4 = {0, 0, 0, 0, 0, 0, 0, 0};
    __builtin_amdgcn_tensor_load_to_lds(g0, g1, g2, g3, g4, 0);
#else
    __builtin_amdgcn_tensor_load_to_lds(g0, g1, g2, g3, 0);
#endif
    __builtin_amdgcn_s_wait_tensorcnt(0);
  }
  if (tid == 64) {
    __builtin_prefetch(Wp, 0, 3);
    __builtin_prefetch(Wp + WP_WL1, 0, 3);
    __builtin_prefetch(Wp + WP_WL2, 0, 3);
  }
  // procedural higher-l w3j tensors (sqrt(2*l3+1) folded in)
  for (int idx = tid; idx < 105 + 125 + 105; idx += 256) {
    if (idx < 105) {
      int i = idx / 15, r = idx % 15, j = r / 3, k = r % 3;
      C321s[idx] = surro(3, 2, 1, i, j, k) * 1.7320508f;
    } else if (idx < 230) {
      int q = idx - 105;
      int i = q / 25, r = q % 25, j = r / 5, k = r % 5;
      C222s[q] = surro(2, 2, 2, i, j, k) * 2.2360680f;
    } else {
      int q = idx - 230;
      int i = q / 15, r = q % 15, j = r / 3, k = r % 5;
      C312s[q] = surro(3, 1, 2, i, j, k) * 2.2360680f;
    }
  }
  __syncthreads();

  // ---- phase 1: per-(edge,channel) CG tensor products -> LDS f16 tiles ----
  for (int p4 = 0; p4 < 4; ++p4) {
    int idx = p4 * 256 + tid;       // 16 edges x 64 channels
    int le = idx >> 6, u = idx & 63;
    int e = e0 + le;
    const float* nb = wYsh + le * 1024 + u * 16;
    float wy[16];
#pragma unroll
    for (int k = 0; k < 16; ++k) wy[k] = nb[k] * EPS_W;
    const float* Vr = V + (size_t)e * 576;
    float V0 = Vr[u];
    float V1[3], V2[5];
#pragma unroll
    for (int j = 0; j < 3; ++j) V1[j] = Vr[64 + u * 3 + j];
#pragma unroll
    for (int j = 0; j < 5; ++j) V2[j] = Vr[256 + u * 5 + j];

    float s1 = (wy[1] * V1[0] + wy[2] * V1[1] + wy[3] * V1[2]) * 0.57735027f;
    float s2 = (wy[4] * V2[0] + wy[5] * V2[1] + wy[6] * V2[2] + wy[7] * V2[3] +
                wy[8] * V2[4]) * 0.44721360f;
    hin[le * 256 + u]       = xh[(size_t)e * 64 + u];
    hin[le * 256 + 64 + u]  = (_Float16)(wy[0] * V0);
    hin[le * 256 + 128 + u] = (_Float16)s1;
    hin[le * 256 + 192 + u] = (_Float16)s2;

#define PW(c, p, val) Psh[((c) * 16 + le) * 320 + (p) * 64 + u] = (_Float16)(val)
#define DW(c, p, val) Dsh[((c) * 16 + le) * 320 + (p) * 64 + u] = (_Float16)(val)
#pragma unroll
    for (int k = 0; k < 3; ++k) PW(k, 0, wy[0] * V1[k]);
#pragma unroll
    for (int k = 0; k < 3; ++k) PW(k, 1, wy[1 + k] * V0);
#pragma unroll
    for (int k = 0; k < 3; ++k) {  // (1,2,1)
      float acc = 0.0f;
#pragma unroll
      for (int i = 0; i < 3; ++i)
#pragma unroll
        for (int j = 0; j < 5; ++j) acc += wy[1 + i] * V2[j] * C112c[i][k][j];
      PW(k, 2, acc * 1.7320508f);
    }
#pragma unroll
    for (int k = 0; k < 3; ++k) {  // (2,1,1)
      float acc = 0.0f;
#pragma unroll
      for (int i = 0; i < 5; ++i)
#pragma unroll
        for (int j = 0; j < 3; ++j) acc += wy[4 + i] * V1[j] * C112c[j][k][i];
      PW(k, 3, acc * 1.7320508f);
    }
#pragma unroll
    for (int k = 0; k < 3; ++k) {  // (3,2,1)
      float acc = 0.0f;
#pragma unroll
      for (int i = 0; i < 7; ++i)
#pragma unroll
        for (int j = 0; j < 5; ++j) acc += wy[9 + i] * V2[j] * C321s[(i * 5 + j) * 3 + k];
      PW(k, 4, acc);
    }
#pragma unroll
    for (int k = 0; k < 5; ++k) DW(k, 0, wy[0] * V2[k]);
#pragma unroll
    for (int k = 0; k < 5; ++k) DW(k, 2, wy[4 + k] * V0);
#pragma unroll
    for (int k = 0; k < 5; ++k) {  // (1,1,2)
      float acc = 0.0f;
#pragma unroll
      for (int i = 0; i < 3; ++i)
#pragma unroll
        for (int j = 0; j < 3; ++j) acc += wy[1 + i] * V1[j] * C112c[i][j][k];
      DW(k, 1, acc * 2.2360680f);
    }
#pragma unroll
    for (int k = 0; k < 5; ++k) {  // (2,2,2)
      float acc = 0.0f;
#pragma unroll
      for (int i = 0; i < 5; ++i)
#pragma unroll
        for (int j = 0; j < 5; ++j) acc += wy[4 + i] * V2[j] * C222s[(i * 5 + j) * 5 + k];
      DW(k, 3, acc);
    }
#pragma unroll
    for (int k = 0; k < 5; ++k) {  // (3,1,2)
      float acc = 0.0f;
#pragma unroll
      for (int i = 0; i < 7; ++i)
#pragma unroll
        for (int j = 0; j < 3; ++j) acc += wy[9 + i] * V1[j] * C312s[(i * 3 + j) * 5 + k];
      DW(k, 4, acc);
    }
#undef PW
#undef DW
    if (u == 0) {  // radial envelope, p=6
      float vx = vectors[e * 3], vy = vectors[e * 3 + 1], vz = vectors[e * 3 + 2];
      float d = sqrtf(vx * vx + vy * vy + vz * vz);
      float d2 = d * d, d6 = d2 * d2 * d2;
      float env = 1.0f - 28.0f * d6 + 48.0f * d6 * d - 21.0f * d6 * d2;
      envsh[le] = (d < 1.0f) ? env : 0.0f;
    }
  }
  __syncthreads();  // also retires all wYsh reads before h1/h2 aliasing writes

  // ---- phase 2: WMMA GEMMs ----
  int wave = tid >> 5, lane = tid & 31;
  if (wave == 0) {
    // MLP: hin(16x256)@W1 -> silu -> @W2 -> silu -> @W3 -> *env
    v8f acc[4] = {zero8(), zero8(), zero8(), zero8()};
    for (int kt = 0; kt < 8; ++kt) {
      v16h a = load_a_c(hin, 256, kt, lane);
#pragma unroll
      for (int t = 0; t < 4; ++t)
        acc[t] = wmma16(a, load_b_pk(Wp + WP_W1, t, kt, lane), acc[t]);
    }
#pragma unroll
    for (int t = 0; t < 4; ++t)
#pragma unroll
      for (int r = 0; r < 8; ++r) {
        int m = r + (lane >> 4) * 8, n = t * 16 + (lane & 15);
        h1[m * 64 + n] = (_Float16)silu(acc[t][r] * 0.0625f);
      }
    wave_sync_lds();
#pragma unroll
    for (int t = 0; t < 4; ++t) acc[t] = zero8();
    for (int kt = 0; kt < 2; ++kt) {
      v16h a = load_a_c(h1, 64, kt, lane);
#pragma unroll
      for (int t = 0; t < 4; ++t)
        acc[t] = wmma16(a, load_b_pk(Wp + WP_W2, t, kt, lane), acc[t]);
    }
#pragma unroll
    for (int t = 0; t < 4; ++t)
#pragma unroll
      for (int r = 0; r < 8; ++r) {
        int m = r + (lane >> 4) * 8, n = t * 16 + (lane & 15);
        h2[m * 64 + n] = (_Float16)silu(acc[t][r] * 0.125f);
      }
    wave_sync_lds();
#pragma unroll
    for (int t = 0; t < 4; ++t) acc[t] = zero8();
    for (int kt = 0; kt < 2; ++kt) {
      v16h a = load_a_c(h2, 64, kt, lane);
#pragma unroll
      for (int t = 0; t < 4; ++t)
        acc[t] = wmma16(a, load_b_pk(Wp + WP_W3, t, kt, lane), acc[t]);
    }
#pragma unroll
    for (int t = 0; t < 4; ++t)
#pragma unroll
      for (int r = 0; r < 8; ++r) {
        int m = r + (lane >> 4) * 8, n = t * 16 + (lane & 15);
        out[(size_t)(e0 + m) * 576 + n] = acc[t][r] * 0.125f * envsh[m];
      }
  } else {
    // 32 independent units: 8 components x 4 N-tiles of (16x320)@(320x64)
    for (int unit = wave - 1; unit < 32; unit += 7) {
      int c8 = unit >> 2, t = unit & 3;
      bool isP = (c8 < 3);
      int comp = isP ? c8 : (c8 - 3);
      const _Float16* Ab = isP ? (Psh + comp * 16 * 320) : (Dsh + comp * 16 * 320);
      const _Float16* Wg = Wp + (isP ? WP_WL1 : WP_WL2);
      v8f acc = zero8();
      for (int kt = 0; kt < 10; ++kt) {
        v16h a = load_a_c(Ab, 320, kt, lane);
        v16h b = load_b_pk(Wg, t, kt, lane);
        acc = wmma16(a, b, acc);
      }
      int n = t * 16 + (lane & 15);
      int col = isP ? (64 + n * 3 + comp) : (256 + n * 5 + comp);
#pragma unroll
      for (int r = 0; r < 8; ++r) {
        int m = r + (lane >> 4) * 8;
        out[(size_t)(e0 + m) * 576 + col] = acc[r] * 0.05590170f;  // 1/sqrt(320)
      }
    }
  }
}

extern "C" void kernel_launch(void* const* d_in, const int* in_sizes, int n_in,
                              void* d_out, int out_size, void* d_ws, size_t ws_size,
                              hipStream_t stream) {
  const float* vectors = (const float*)d_in[0];
  const float* x       = (const float*)d_in[1];
  const float* V       = (const float*)d_in[2];
  const int*   senders = (const int*)d_in[3];
  const float* Ww      = (const float*)d_in[4];
  const float* W1      = (const float*)d_in[5];
  const float* W2      = (const float*)d_in[6];
  const float* W3      = (const float*)d_in[7];
  const float* Wl1     = (const float*)d_in[8];
  const float* Wl2     = (const float*)d_in[9];
  int E = in_sizes[0] / 3;

  float*    nodeAcc = (float*)d_ws;
  _Float16* Wp      = (_Float16*)((char*)d_ws + WS_WP_OFF);
  _Float16* xh      = (_Float16*)((char*)d_ws + WS_XH_OFF);

  hipMemsetAsync(nodeAcc, 0, WS_NODE_BYTES, stream);
  pack_weights_kernel<<<(WP_TOTAL + 255) / 256, 256, 0, stream>>>(W1, W2, W3, Wl1, Wl2, Ww, Wp);
  pack_x_kernel<<<(E * 64) / 256, 256, 0, stream>>>(x, xh, E * 64);
  sh_scatter_kernel<<<E / 16, 256, 0, stream>>>(vectors, xh, senders, Wp, nodeAcc, E);
  allegro_main_kernel<<<E / 16, 256, 0, stream>>>(vectors, xh, V, senders, Wp, nodeAcc,
                                                  (float*)d_out, E);
}
